// SelfAttention_12713103196701
// MI455X (gfx1250) — matrix operand order repaired and checked by reading, not compiled, not run on hardware
//
#include <hip/hip_runtime.h>
#include <hip/hip_bf16.h>

typedef __attribute__((ext_vector_type(16))) __bf16 v16bf;
typedef __attribute__((ext_vector_type(8)))  float  v8f;

#define T_SEQ   2048
#define DEMB    1024
#define DQKV    64
#define NBATCH  8
#define ATT_SCALE 0.125f     // 1/sqrt(64)
#define WAVES   4            // waves per workgroup (wave32)

union BF16Frag {
    unsigned short u[16];
    __bf16         b[16];
    v16bf          v;
};

// float -> bf16 bits via hardware cvt (RNE)
__device__ __forceinline__ unsigned short bf_bits(float f) {
    __bf16 h = (__bf16)f;
    return __builtin_bit_cast(unsigned short, h);
}

// copy 8 bf16 (16 bytes) — lowers to global/ds b128 loads
__device__ __forceinline__ void load8(unsigned short* dst, const unsigned short* src) {
    *reinterpret_cast<uint4*>(dst) = *reinterpret_cast<const uint4*>(src);
}

__device__ __forceinline__ float redmax16(float v) {
    v = fmaxf(v, __shfl_xor(v, 1));
    v = fmaxf(v, __shfl_xor(v, 2));
    v = fmaxf(v, __shfl_xor(v, 4));
    v = fmaxf(v, __shfl_xor(v, 8));
    return v;
}
__device__ __forceinline__ float redsum16(float v) {
    v += __shfl_xor(v, 1);
    v += __shfl_xor(v, 2);
    v += __shfl_xor(v, 4);
    v += __shfl_xor(v, 8);
    return v;
}

// ---------------------------------------------------------------------------
// Kernel 0: convert Wq/Wk/Wv (fp32 [1024,64]) -> bf16 transposed Wt[3][64][1024]
// ---------------------------------------------------------------------------
__global__ void wprep_kernel(const float* __restrict__ Wq,
                             const float* __restrict__ Wk,
                             const float* __restrict__ Wv,
                             unsigned short* __restrict__ Wt) {
    int idx = blockIdx.x * blockDim.x + threadIdx.x;
    if (idx >= 3 * DQKV * DEMB) return;
    int m   = idx >> 16;        // / (64*1024)
    int rem = idx & 65535;
    int n   = rem >> 10;        // / 1024
    int k   = rem & 1023;
    const float* W = (m == 0) ? Wq : (m == 1) ? Wk : Wv;
    Wt[idx] = bf_bits(W[k * DQKV + n]);
}

// ---------------------------------------------------------------------------
// Kernel 1: QKV projection with async-LDS double-buffered weight staging.
// Block = 4 waves; each wave computes a 16-token x 64-feature tile of Q,K,V.
// Per K-step the block cooperatively stages a 192x32 bf16 weight chunk (12 KB)
// into LDS with global_load_async_to_lds_b128 (ASYNCcnt).
// ---------------------------------------------------------------------------
#define WROWS   192              // 3 matrices * 64 output cols
#define WCHUNK  (WROWS * 32)     // ushorts per staged chunk

__device__ __forceinline__ void stage_weights(const unsigned short* __restrict__ Wt,
                                              int kk, unsigned short* dstbuf, int tid) {
    // 192 rows * 64 bytes = 768 b128 chunks; 128 threads * 6 iters
    #pragma unroll
    for (int it = 0; it < 6; ++it) {
        const int ci   = tid + 128 * it;
        const int r    = ci >> 2;          // weight row 0..191
        const int part = ci & 3;           // 8-ushort sub-chunk
        unsigned lds_addr = (unsigned)(uintptr_t)(dstbuf + r * 32 + part * 8);
        unsigned long long gaddr =
            (unsigned long long)(uintptr_t)(Wt + (size_t)r * DEMB + kk + part * 8);
        asm volatile("global_load_async_to_lds_b128 %0, %1, off"
                     :: "v"(lds_addr), "v"(gaddr) : "memory");
    }
}

__global__ void __launch_bounds__(WAVES * 32)
qkv_proj_kernel(const float* __restrict__ x, const unsigned short* __restrict__ Wt,
                unsigned short* __restrict__ Qw, unsigned short* __restrict__ Kw,
                unsigned short* __restrict__ Vt) {
    __shared__ alignas(16) unsigned short wlds[2][WCHUNK];   // 24 KB double buffer

    const int tid  = threadIdx.x;
    const int lane = tid & 31;
    const int wid  = tid >> 5;
    const int cL   = lane & 15;
    const int h    = lane >> 4;
    const size_t t0 = ((size_t)blockIdx.x * WAVES + wid) * 16;   // flat B*T row

    const v8f vzero = {};
    v8f acc[3][4];
    #pragma unroll
    for (int m = 0; m < 3; ++m)
        #pragma unroll
        for (int nt = 0; nt < 4; ++nt) acc[m][nt] = vzero;

    const float* xr = x + (t0 + (size_t)cL) * DEMB;   // A-frag row = lane%16

    // software-pipelined x loads (A fragment source)
    float4 fa0, fa1, fa2, fa3;
    auto loadA = [&](int kk) {
        fa0 = *reinterpret_cast<const float4*>(xr + kk + h * 8);
        fa1 = *reinterpret_cast<const float4*>(xr + kk + h * 8 + 4);
        fa2 = *reinterpret_cast<const float4*>(xr + kk + 16 + h * 8);
        fa3 = *reinterpret_cast<const float4*>(xr + kk + 16 + h * 8 + 4);
    };
    loadA(0);
    stage_weights(Wt, 0, &wlds[0][0], tid);   // prologue: buffer 0 in flight

    int cur = 0;
    for (int kk = 0; kk < DEMB; kk += 32) {
        const bool has_next = (kk + 32) < DEMB;
        if (has_next) stage_weights(Wt, kk + 32, &wlds[cur ^ 1][0], tid);

        // convert current A chunk: element j -> K = (j/8)*16 + h*8 + j%8
        BF16Frag a;
        a.b[0]  = (__bf16)fa0.x; a.b[1]  = (__bf16)fa0.y;
        a.b[2]  = (__bf16)fa0.z; a.b[3]  = (__bf16)fa0.w;
        a.b[4]  = (__bf16)fa1.x; a.b[5]  = (__bf16)fa1.y;
        a.b[6]  = (__bf16)fa1.z; a.b[7]  = (__bf16)fa1.w;
        a.b[8]  = (__bf16)fa2.x; a.b[9]  = (__bf16)fa2.y;
        a.b[10] = (__bf16)fa2.z; a.b[11] = (__bf16)fa2.w;
        a.b[12] = (__bf16)fa3.x; a.b[13] = (__bf16)fa3.y;
        a.b[14] = (__bf16)fa3.z; a.b[15] = (__bf16)fa3.w;
        if (has_next) loadA(kk + 32);      // overlap next x loads with WMMAs

        // wait for CURRENT buffer's 6 asyncs (newest 6 stay in flight), then sync
        if (has_next) asm volatile("s_wait_asynccnt 0x6" ::: "memory");
        else          asm volatile("s_wait_asynccnt 0x0" ::: "memory");
        __syncthreads();

        const unsigned short* wb = &wlds[cur][0];
        #pragma unroll
        for (int m = 0; m < 3; ++m) {
            #pragma unroll
            for (int nt = 0; nt < 4; ++nt) {
                // B-frag: col = lane%16, element j -> K = (j/8)*16 + h*8 + j%8
                BF16Frag bfr;
                const unsigned short* wp = wb + ((m * 4 + nt) * 16 + cL) * 32 + h * 8;
                load8(bfr.u + 0, wp);
                load8(bfr.u + 8, wp + 16);
                acc[m][nt] = __builtin_amdgcn_wmma_f32_16x16x32_bf16(
                    false, a.v, false, bfr.v, (short)0, acc[m][nt], false, false);
            }
        }
        __syncthreads();   // nobody overwrites 'cur^1'->'cur' before all waves read
        cur ^= 1;
    }

    // C-tile: row = h*8 + i, col = nt*16 + cL
    #pragma unroll
    for (int nt = 0; nt < 4; ++nt) {
        #pragma unroll
        for (int i = 0; i < 8; ++i) {
            const size_t ft  = t0 + h * 8 + i;
            const int    col = nt * 16 + cL;
            Qw[ft * DQKV + col] = bf_bits(acc[0][nt][i]);
            Kw[ft * DQKV + col] = bf_bits(acc[1][nt][i]);
            const size_t b = ft / T_SEQ, tt = ft % T_SEQ;
            Vt[(b * DQKV + col) * T_SEQ + tt] = bf_bits(acc[2][nt][i]);   // V transposed
        }
    }
}

// ---------------------------------------------------------------------------
// Kernel 2: causal flash attention. One wave per 16-query tile, 32-key blocks.
// V fragments hoisted above softmax; next block prefetched (global_prefetch_b8).
// ---------------------------------------------------------------------------
__global__ void __launch_bounds__(WAVES * 32)
attn_kernel(const unsigned short* __restrict__ Qw, const unsigned short* __restrict__ Kw,
            const unsigned short* __restrict__ Vt, float* __restrict__ out) {
    __shared__ alignas(16) unsigned short pls[WAVES][16 * 32];   // P staging per wave

    const int lane = threadIdx.x & 31;
    const int wid  = threadIdx.x >> 5;
    const int cL   = lane & 15;
    const int h    = lane >> 4;
    const size_t q0f = ((size_t)blockIdx.x * WAVES + wid) * 16;  // flat query row
    const int b  = (int)(q0f / T_SEQ);
    const int q0 = (int)(q0f % T_SEQ);

    const unsigned short* Kb = Kw + (size_t)b * T_SEQ * DQKV;
    const unsigned short* Vb = Vt + (size_t)b * DQKV * T_SEQ;

    // Q A-fragments for d=[0,32) and d=[32,64): row = lane%16
    BF16Frag qa0, qa1;
    {
        const unsigned short* qr = Qw + (q0f + (size_t)cL) * DQKV;
        load8(qa0.u + 0, qr +  0 + h * 8);
        load8(qa0.u + 8, qr + 16 + h * 8);
        load8(qa1.u + 0, qr + 32 + h * 8);
        load8(qa1.u + 8, qr + 48 + h * 8);
    }

    const v8f vzero = {};
    v8f o[4];
    #pragma unroll
    for (int nt = 0; nt < 4; ++nt) o[nt] = vzero;
    float mrow[8], lrow[8];
    #pragma unroll
    for (int i = 0; i < 8; ++i) { mrow[i] = -3.0e38f; lrow[i] = 0.0f; }

    unsigned short* prow = &pls[wid][0];
    const int nkb = (q0 + 16 + 31) >> 5;     // causal: keys [0, q0+16)

    for (int blk = 0; blk < nkb; ++blk) {
        const int j0 = blk * 32;

        // prefetch next key block's K/V cachelines
        if (blk + 1 < nkb) {
            const int j1 = j0 + 32;
            __builtin_prefetch(Kb + (size_t)(j1 + lane) * DQKV, 0, 0);
            __builtin_prefetch(Vb + (size_t)lane * T_SEQ + j1, 0, 0);
            __builtin_prefetch(Vb + (size_t)(32 + lane) * T_SEQ + j1, 0, 0);
        }

        // ---- K B-fragments, then V B-fragments (latency hidden under S+softmax) ----
        BF16Frag kb0[2], kb1[2], vb[4];
        #pragma unroll
        for (int st = 0; st < 2; ++st) {
            const unsigned short* kr = Kb + (size_t)(j0 + st * 16 + cL) * DQKV;
            load8(kb0[st].u + 0, kr +  0 + h * 8);
            load8(kb0[st].u + 8, kr + 16 + h * 8);
            load8(kb1[st].u + 0, kr + 32 + h * 8);
            load8(kb1[st].u + 8, kr + 48 + h * 8);
        }
        #pragma unroll
        for (int nt = 0; nt < 4; ++nt) {
            const unsigned short* vr = Vb + (size_t)(nt * 16 + cL) * T_SEQ + j0 + h * 8;
            load8(vb[nt].u + 0, vr);
            load8(vb[nt].u + 8, vr + 16);
        }

        // ---- S = Q · K^T  (two 16x16 tiles, K-dim 64 split 32+32) ----
        v8f s[2];
        #pragma unroll
        for (int st = 0; st < 2; ++st) {
            v8f sa = vzero;
            sa = __builtin_amdgcn_wmma_f32_16x16x32_bf16(false, qa0.v, false, kb0[st].v,
                                                         (short)0, sa, false, false);
            sa = __builtin_amdgcn_wmma_f32_16x16x32_bf16(false, qa1.v, false, kb1[st].v,
                                                         (short)0, sa, false, false);
            s[st] = sa;
        }

        // ---- scale + causal mask + online softmax (rows live in lane-halves) ----
        #pragma unroll
        for (int i = 0; i < 8; ++i) {
            const int qrow = q0 + h * 8 + i;
            float v0 = s[0][i] * ATT_SCALE;
            float v1 = s[1][i] * ATT_SCALE;
            if (j0 + cL      > qrow) v0 = -3.0e38f;
            if (j0 + 16 + cL > qrow) v1 = -3.0e38f;
            const float rm    = redmax16(fmaxf(v0, v1));
            const float mnew  = fmaxf(mrow[i], rm);
            const float alpha = __expf(mrow[i] - mnew);
            const float p0 = __expf(v0 - mnew);
            const float p1 = __expf(v1 - mnew);
            const float rs = redsum16(p0 + p1);
            lrow[i] = lrow[i] * alpha + rs;
            mrow[i] = mnew;
            #pragma unroll
            for (int nt = 0; nt < 4; ++nt) o[nt][i] = o[nt][i] * alpha;
            // stage P (C-layout -> LDS row-major 16x32)
            prow[(h * 8 + i) * 32 +      cL] = bf_bits(p0);
            prow[(h * 8 + i) * 32 + 16 + cL] = bf_bits(p1);
        }

        // ---- reload P as A-fragment (ds_load_b128; same-wave DS order + waits) ----
        BF16Frag pa;
        load8(pa.u + 0, prow + cL * 32 +      h * 8);
        load8(pa.u + 8, prow + cL * 32 + 16 + h * 8);

        // ---- O += P · V  (V already resident in registers) ----
        #pragma unroll
        for (int nt = 0; nt < 4; ++nt) {
            o[nt] = __builtin_amdgcn_wmma_f32_16x16x32_bf16(
                false, pa.v, false, vb[nt].v, (short)0, o[nt], false, false);
        }
    }

    // ---- normalize and store fp32 output ----
    #pragma unroll
    for (int nt = 0; nt < 4; ++nt) {
        #pragma unroll
        for (int i = 0; i < 8; ++i) {
            const float inv = 1.0f / lrow[i];
            out[(q0f + h * 8 + i) * DQKV + nt * 16 + cL] = o[nt][i] * inv;
        }
    }
}

// ---------------------------------------------------------------------------
extern "C" void kernel_launch(void* const* d_in, const int* in_sizes, int n_in,
                              void* d_out, int out_size, void* d_ws, size_t ws_size,
                              hipStream_t stream) {
    (void)in_sizes; (void)n_in; (void)out_size; (void)ws_size;
    const float* x  = (const float*)d_in[0];
    const float* Wq = (const float*)d_in[1];
    const float* Wk = (const float*)d_in[2];
    const float* Wv = (const float*)d_in[3];
    float* out = (float*)d_out;

    unsigned short* ws = (unsigned short*)d_ws;
    unsigned short* Wt = ws;                                   // 3*64*1024 bf16
    unsigned short* Qw = Wt + (size_t)3 * DQKV * DEMB;         // B*T*64 bf16
    unsigned short* Kw = Qw + (size_t)NBATCH * T_SEQ * DQKV;   // B*T*64 bf16
    unsigned short* Vt = Kw + (size_t)NBATCH * T_SEQ * DQKV;   // B*64*T bf16 (transposed)

    {
        const int n = 3 * DQKV * DEMB;
        wprep_kernel<<<(n + 255) / 256, 256, 0, stream>>>(Wq, Wk, Wv, Wt);
    }
    const int tiles = NBATCH * T_SEQ / 16;        // 1024 tiles
    qkv_proj_kernel<<<tiles / WAVES, WAVES * 32, 0, stream>>>(x, Wt, Qw, Kw, Vt);
    attn_kernel<<<tiles / WAVES, WAVES * 32, 0, stream>>>(Qw, Kw, Vt, out);
}
